// HierarchicalTemporalAggregation_50749333570250
// MI455X (gfx1250) — matrix operand order repaired and checked by reading, not compiled, not run on hardware
//
#include <hip/hip_runtime.h>

// ---------------- problem constants ----------------
constexpr int Bc  = 8;
constexpr int Sc  = 2048;
constexpr int Fc  = 512;
constexpr int Hc  = 4;
constexpr int Wc  = 32;          // local window
constexpr int DHc = 128;         // head dim
constexpr int F3c = 3 * Fc;      // 1536
constexpr int BSc = Bc * Sc;     // 16384 rows
constexpr float EPSc = 1e-5f;

typedef __attribute__((ext_vector_type(16))) _Float16 v16h;
typedef __attribute__((ext_vector_type(8)))  float    v8f;

// ---------------- WMMA fragment helpers (contiguous k order k = 16*(lane/16) + s, same for A and B) ----------------
// A fragment: 16x32 (M x K) f16. lanes 0-15: rows 0-15, K 0..15;
// lanes 16-31: rows 0-15, K 16..31.
__device__ __forceinline__ v16h load_a_frag(const _Float16* s, int ld) {
  int lane = threadIdx.x & 31;
  int row  = lane & 15;
  int kb   = (lane & 16) ? 16 : 0;
  const _Float16* r = s + (size_t)row * ld;
  v16h a;
#pragma unroll
  for (int p = 0; p < 4; ++p) {
    a[2 * p]         = r[kb + 2 * p];
    a[2 * p + 1]     = r[kb + 2 * p + 1];
    a[8 + 2 * p]     = r[8 + kb + 2 * p];
    a[8 + 2 * p + 1] = r[8 + kb + 2 * p + 1];
  }
  return a;
}

// B fragment 32x16 (K x N) where source is stored [n][k] row-major (e.g. weights [N,K]).
__device__ __forceinline__ v16h load_b_frag_nk(const _Float16* s, int ld) {
  int lane = threadIdx.x & 31;
  int n    = lane & 15;
  int kb   = (lane & 16) ? 16 : 0;
  const _Float16* r = s + (size_t)n * ld;
  v16h b;
#pragma unroll
  for (int j = 0; j < 8; ++j) {
    b[2 * j]     = r[kb + 2 * j];
    b[2 * j + 1] = r[kb + 2 * j + 1];
  }
  return b;
}

// B fragment 32x16 (K x N) where source is stored [k][n] row-major (e.g. V chunk [key][d]).
__device__ __forceinline__ v16h load_b_frag_kn(const _Float16* s, int ld) {
  int lane = threadIdx.x & 31;
  int n    = lane & 15;
  int kb   = (lane & 16) ? 16 : 0;
  v16h b;
#pragma unroll
  for (int j = 0; j < 8; ++j) {
    b[2 * j]     = s[(size_t)(kb + 2 * j) * ld + n];
    b[2 * j + 1] = s[(size_t)(kb + 2 * j + 1) * ld + n];
  }
  return b;
}

// ---------------- f32 -> f16 convert ----------------
__global__ void cvt_f16_kernel(const float* __restrict__ src, _Float16* __restrict__ dst, int n) {
  int i = blockIdx.x * blockDim.x + threadIdx.x;
  if (i < n) dst[i] = (_Float16)src[i];
}

// ---------------- GEMM: C[M,N] = A[M,K](f16) * W[N,K]^T (f16) + bias ----------------
// block = 128 threads (4 waves); block tile 128x64; wave wv owns rows [32wv, 32wv+32)
// (2 A-fragments) x 64 cols (4 B-fragments) -> 8 WMMA per k-step with B reuse.
// TOF32: write f32 C (ld = N); else write f16 C (ld = ldch).
template <bool TOF32>
__global__ __launch_bounds__(128) void gemm_f16_kernel(
    const _Float16* __restrict__ A, const _Float16* __restrict__ Wt,
    const float* __restrict__ bias,
    float* __restrict__ Cf, _Float16* __restrict__ Ch,
    int M, int N, int K, int ldch) {
  __shared__ __align__(16) _Float16 As[128][40];
  __shared__ __align__(16) _Float16 Bs[64][40];

  int tid = threadIdx.x;
  int wv  = tid >> 5;
  int m0  = blockIdx.y * 128;
  int n0  = blockIdx.x * 64;

  v8f acc[2][4] = {};

  int browB = tid >> 1;             // 0..63
  int bcofB = (tid & 1) << 4;       // 0 or 16 halves

  for (int k0 = 0; k0 < K; k0 += 32) {
    // A: each thread loads one 32-half row; B: 16 halves.
    const uint4* ga = (const uint4*)(A  + (size_t)(m0 + tid) * K + k0);
    uint4 a0 = ga[0], a1 = ga[1], a2 = ga[2], a3 = ga[3];
    const uint4* gw = (const uint4*)(Wt + (size_t)(n0 + browB) * K + k0 + bcofB);
    uint4 w0 = gw[0], w1 = gw[1];
    if (k0 + 32 < K) {   // prefetch next k-chunk (global_prefetch_b8)
      __builtin_prefetch(A  + (size_t)(m0 + tid) * K + k0 + 32, 0, 0);
      __builtin_prefetch(Wt + (size_t)(n0 + browB) * K + k0 + 32, 0, 0);
    }
    __syncthreads();
    *(uint4*)&As[tid][0]  = a0;
    *(uint4*)&As[tid][8]  = a1;
    *(uint4*)&As[tid][16] = a2;
    *(uint4*)&As[tid][24] = a3;
    *(uint4*)&Bs[browB][bcofB]     = w0;
    *(uint4*)&Bs[browB][bcofB + 8] = w1;
    __syncthreads();

    v16h af0 = load_a_frag(&As[wv * 32][0], 40);
    v16h af1 = load_a_frag(&As[wv * 32 + 16][0], 40);
#pragma unroll
    for (int c = 0; c < 4; ++c) {
      v16h bf = load_b_frag_nk(&Bs[c * 16][0], 40);
      acc[0][c] = __builtin_amdgcn_wmma_f32_16x16x32_f16(
          false, af0, false, bf, (short)0, acc[0][c], false, false);
      acc[1][c] = __builtin_amdgcn_wmma_f32_16x16x32_f16(
          false, af1, false, bf, (short)0, acc[1][c], false, false);
    }
  }

  // epilogue: C 16x16 layout -> lane (n=lane&15), vgpr r -> row r (+8 for lanes 16-31)
  int lane = tid & 31;
  int nn   = lane & 15;
  int rhi  = (lane & 16) ? 8 : 0;
  float bc[4];
#pragma unroll
  for (int c = 0; c < 4; ++c) bc[c] = bias[n0 + c * 16 + nn];
#pragma unroll
  for (int mi = 0; mi < 2; ++mi) {
#pragma unroll
    for (int c = 0; c < 4; ++c) {
      int col = n0 + c * 16 + nn;
#pragma unroll
      for (int r = 0; r < 8; ++r) {
        int row = m0 + wv * 32 + mi * 16 + r + rhi;
        float v = acc[mi][c][r] + bc[c];
        if (TOF32) Cf[(size_t)row * N + col] = v;
        else       Ch[(size_t)row * ldch + col] = (_Float16)v;
      }
    }
  }
}

// ---------------- Flash attention (global or windowed) ----------------
// qkv: [BS, 3F] f16 rows (q | k | v).  attn: [BS, F] f16.
// One wave handles 16 q-rows of one (b,h). window==0 -> full sequence.
#define FW_WAVES 2
__global__ __launch_bounds__(32 * FW_WAVES) void flash_kernel(
    const _Float16* __restrict__ qkv, _Float16* __restrict__ attn, int window) {
  __shared__ __align__(16) _Float16 Kl[FW_WAVES][32 * DHc];
  __shared__ __align__(16) _Float16 Vl[FW_WAVES][32 * DHc];
  __shared__ __align__(16) _Float16 Pl[FW_WAVES][16 * 32];

  int wv   = threadIdx.x >> 5;
  int lane = threadIdx.x & 31;
  int qt   = blockIdx.x * FW_WAVES + wv;
  const int tiles_per_bh = Sc / 16;
  int bh = qt / tiles_per_bh;
  int ti = qt % tiles_per_bh;
  int b  = bh / Hc;
  int h  = bh % Hc;
  int s0 = ti * 16;

  int kv0   = window ? (s0 / window) * window : 0;
  int kvlen = window ? window : Sc;
  const float scale = 0.0883883476483184f;   // 1/sqrt(128)

  // Q fragments: 4 x (16x32) covering DH=128
  const _Float16* qbase = qkv + (size_t)(b * Sc + s0) * F3c + h * DHc;
  v16h qf[4];
#pragma unroll
  for (int d = 0; d < 4; ++d) qf[d] = load_a_frag(qbase + d * 32, F3c);

  v8f o[8] = {};                 // 16 rows x 128 cols accumulator (8 d-tiles)
  float mrow[8], lrow[8];
#pragma unroll
  for (int r = 0; r < 8; ++r) { mrow[r] = -1e30f; lrow[r] = 0.0f; }

  int nn  = lane & 15;
  int rhi = (lane & 16) ? 8 : 0;

  for (int kc = 0; kc < kvlen; kc += 32) {
    // cooperative per-wave load of 32 K rows and 32 V rows (128 halves each)
    int key = kv0 + kc + lane;
    const uint4* ks = (const uint4*)(qkv + (size_t)(b * Sc + key) * F3c + Fc + h * DHc);
    const uint4* vs = (const uint4*)(qkv + (size_t)(b * Sc + key) * F3c + 2 * Fc + h * DHc);
    uint4* kd = (uint4*)&Kl[wv][lane * DHc];
    uint4* vd = (uint4*)&Vl[wv][lane * DHc];
#pragma unroll
    for (int i = 0; i < 16; ++i) { kd[i] = ks[i]; vd[i] = vs[i]; }
    if (kc + 32 < kvlen) {       // prefetch next chunk's K/V rows
      __builtin_prefetch(qkv + (size_t)(b * Sc + key + 32) * F3c + Fc + h * DHc, 0, 0);
      __builtin_prefetch(qkv + (size_t)(b * Sc + key + 32) * F3c + 2 * Fc + h * DHc, 0, 0);
    }

    // scores: 16 q-rows x 32 keys  (two 16-key subtiles, 4 d-steps each)
    v8f sc0 = {}, sc1 = {};
#pragma unroll
    for (int d = 0; d < 4; ++d) {
      v16h bf0 = load_b_frag_nk(&Kl[wv][0 * 16 * DHc + d * 32], DHc);
      sc0 = __builtin_amdgcn_wmma_f32_16x16x32_f16(false, qf[d], false, bf0, (short)0, sc0, false, false);
      v16h bf1 = load_b_frag_nk(&Kl[wv][1 * 16 * DHc + d * 32], DHc);
      sc1 = __builtin_amdgcn_wmma_f32_16x16x32_f16(false, qf[d], false, bf1, (short)0, sc1, false, false);
    }

    // online softmax update per row (rows live across a 16-lane group)
#pragma unroll
    for (int r = 0; r < 8; ++r) {
      float s0v = sc0[r] * scale;
      float s1v = sc1[r] * scale;
      float mx = fmaxf(s0v, s1v);
#pragma unroll
      for (int off = 8; off >= 1; off >>= 1)
        mx = fmaxf(mx, __shfl_xor(mx, off, 32));
      float mn   = fmaxf(mrow[r], mx);
      float corr = __expf(mrow[r] - mn);
      s0v = __expf(s0v - mn);
      s1v = __expf(s1v - mn);
      float sm = s0v + s1v;
#pragma unroll
      for (int off = 8; off >= 1; off >>= 1)
        sm += __shfl_xor(sm, off, 32);
      lrow[r] = lrow[r] * corr + sm;
      mrow[r] = mn;
#pragma unroll
      for (int t = 0; t < 8; ++t) o[t][r] *= corr;
      Pl[wv][(r + rhi) * 32 + nn]      = (_Float16)s0v;
      Pl[wv][(r + rhi) * 32 + 16 + nn] = (_Float16)s1v;
    }

    // O += P(16x32) * V(32x128)
    v16h pf = load_a_frag(&Pl[wv][0], 32);
#pragma unroll
    for (int t = 0; t < 8; ++t) {
      v16h bf = load_b_frag_kn(&Vl[wv][t * 16], DHc);
      o[t] = __builtin_amdgcn_wmma_f32_16x16x32_f16(false, pf, false, bf, (short)0, o[t], false, false);
    }
  }

  // normalize and store
#pragma unroll
  for (int r = 0; r < 8; ++r) {
    float inv = 1.0f / lrow[r];
    int row = b * Sc + s0 + r + rhi;
#pragma unroll
    for (int t = 0; t < 8; ++t)
      attn[(size_t)row * Fc + h * DHc + t * 16 + nn] = (_Float16)(o[t][r] * inv);
  }
}

// ---------------- block reductions ----------------
__device__ __forceinline__ void block_sum2(float a, float b, float& ra, float& rb) {
  __shared__ float sa[8], sb[8];
#pragma unroll
  for (int off = 16; off >= 1; off >>= 1) {
    a += __shfl_xor(a, off, 32);
    b += __shfl_xor(b, off, 32);
  }
  if ((threadIdx.x & 31) == 0) { sa[threadIdx.x >> 5] = a; sb[threadIdx.x >> 5] = b; }
  __syncthreads();
  if (threadIdx.x == 0) {
    float ta = 0.f, tb = 0.f;
#pragma unroll
    for (int i = 0; i < 8; ++i) { ta += sa[i]; tb += sb[i]; }
    sa[0] = ta; sb[0] = tb;
  }
  __syncthreads();
  ra = sa[0]; rb = sb[0];
}

// ---------------- residual + LayerNorm (writes f32 + f16 slice of gate input) ----------------
__global__ __launch_bounds__(256) void add_ln_kernel(
    const float* __restrict__ xin, const float* __restrict__ yin,
    const float* __restrict__ w, const float* __restrict__ bp,
    float* __restrict__ outf, _Float16* __restrict__ outh, int ldh) {
  int row = blockIdx.x, tid = threadIdx.x;
  size_t base = (size_t)row * Fc;
  float v0 = xin[base + tid]       + yin[base + tid];
  float v1 = xin[base + tid + 256] + yin[base + tid + 256];
  float s, ss;
  block_sum2(v0 + v1, v0 * v0 + v1 * v1, s, ss);
  float mu  = s * (1.0f / Fc);
  float var = ss * (1.0f / Fc) - mu * mu;
  float rs  = rsqrtf(var + EPSc);
  float o0 = (v0 - mu) * rs * w[tid]       + bp[tid];
  float o1 = (v1 - mu) * rs * w[tid + 256] + bp[tid + 256];
  outf[base + tid]       = o0;
  outf[base + tid + 256] = o1;
  size_t hb = (size_t)row * ldh;
  outh[hb + tid]       = (_Float16)o0;
  outh[hb + tid + 256] = (_Float16)o1;
}

// ---------------- gate LN + sigmoid + fuse ----------------
__global__ __launch_bounds__(256) void gate_fuse_kernel(
    const float* __restrict__ glin,
    const float* __restrict__ gw, const float* __restrict__ gb,
    const float* __restrict__ lf, const float* __restrict__ gf,
    float* __restrict__ out) {
  int row = blockIdx.x, tid = threadIdx.x;
  size_t base = (size_t)row * Fc;
  float v0 = glin[base + tid];
  float v1 = glin[base + tid + 256];
  float s, ss;
  block_sum2(v0 + v1, v0 * v0 + v1 * v1, s, ss);
  float mu  = s * (1.0f / Fc);
  float var = ss * (1.0f / Fc) - mu * mu;
  float rs  = rsqrtf(var + EPSc);
  float g0 = 1.0f / (1.0f + __expf(-((v0 - mu) * rs * gw[tid] + gb[tid])));
  float g1 = 1.0f / (1.0f + __expf(-((v1 - mu) * rs * gw[tid + 256] + gb[tid + 256])));
  out[base + tid]       = g0 * lf[base + tid]       + (1.0f - g0) * gf[base + tid];
  out[base + tid + 256] = g1 * lf[base + tid + 256] + (1.0f - g1) * gf[base + tid + 256];
}

// ---------------- host-side launch ----------------
extern "C" void kernel_launch(void* const* d_in, const int* in_sizes, int n_in,
                              void* d_out, int out_size, void* d_ws, size_t ws_size,
                              hipStream_t stream) {
  (void)in_sizes; (void)n_in; (void)out_size; (void)ws_size;
  const float* x       = (const float*)d_in[0];
  const float* l_in_w  = (const float*)d_in[1];
  const float* l_in_b  = (const float*)d_in[2];
  const float* l_out_w = (const float*)d_in[3];
  const float* l_out_b = (const float*)d_in[4];
  const float* g_in_w  = (const float*)d_in[5];
  const float* g_in_b  = (const float*)d_in[6];
  const float* g_out_w = (const float*)d_in[7];
  const float* g_out_b = (const float*)d_in[8];
  const float* n1_w    = (const float*)d_in[9];
  const float* n1_b    = (const float*)d_in[10];
  const float* n2_w    = (const float*)d_in[11];
  const float* n2_b    = (const float*)d_in[12];
  const float* gate_w  = (const float*)d_in[13];
  const float* gate_b  = (const float*)d_in[14];
  const float* gln_w   = (const float*)d_in[15];
  const float* gln_b   = (const float*)d_in[16];

  char* ws = (char*)d_ws;
  size_t off = 0;
  auto take = [&](size_t bytes) -> char* {
    char* p = ws + off;
    off += (bytes + 255) & ~(size_t)255;
    return p;
  };
  _Float16* xh    = (_Float16*)take((size_t)BSc * Fc * 2);
  _Float16* qkvh  = (_Float16*)take((size_t)BSc * F3c * 2);
  _Float16* attnh = (_Float16*)take((size_t)BSc * Fc * 2);
  float*    projf = (float*)   take((size_t)BSc * Fc * 4);
  float*    localf= (float*)   take((size_t)BSc * Fc * 4);
  float*    globf = (float*)   take((size_t)BSc * Fc * 4);
  _Float16* gih   = (_Float16*)take((size_t)BSc * 2 * Fc * 2);
  float*    glinf = (float*)   take((size_t)BSc * Fc * 4);
  _Float16* winh  = (_Float16*)take((size_t)F3c * Fc * 2);
  _Float16* wouth = (_Float16*)take((size_t)Fc * Fc * 2);
  _Float16* gwh   = (_Float16*)take((size_t)Fc * 2 * Fc * 2);

  const int TILES = Bc * Hc * (Sc / 16);   // 4096
  dim3 cb(256);
  dim3 gb(128);

  // convert x and (local) weights to f16
  cvt_f16_kernel<<<(BSc * Fc + 255) / 256, cb, 0, stream>>>(x, xh, BSc * Fc);
  cvt_f16_kernel<<<(F3c * Fc + 255) / 256, cb, 0, stream>>>(l_in_w, winh, F3c * Fc);
  cvt_f16_kernel<<<(Fc * Fc + 255) / 256, cb, 0, stream>>>(l_out_w, wouth, Fc * Fc);
  cvt_f16_kernel<<<(Fc * 2 * Fc + 255) / 256, cb, 0, stream>>>(gate_w, gwh, Fc * 2 * Fc);

  // ---- local branch ----
  gemm_f16_kernel<false><<<dim3(F3c / 64, BSc / 128), gb, 0, stream>>>(
      xh, winh, l_in_b, nullptr, qkvh, BSc, F3c, Fc, F3c);
  flash_kernel<<<dim3(TILES / FW_WAVES), dim3(32 * FW_WAVES), 0, stream>>>(qkvh, attnh, Wc);
  gemm_f16_kernel<true><<<dim3(Fc / 64, BSc / 128), gb, 0, stream>>>(
      attnh, wouth, l_out_b, projf, nullptr, BSc, Fc, Fc, 0);
  add_ln_kernel<<<BSc, cb, 0, stream>>>(x, projf, n1_w, n1_b, localf, gih, 2 * Fc);

  // ---- global branch (reuse weight/qkv/attn/proj buffers) ----
  cvt_f16_kernel<<<(F3c * Fc + 255) / 256, cb, 0, stream>>>(g_in_w, winh, F3c * Fc);
  cvt_f16_kernel<<<(Fc * Fc + 255) / 256, cb, 0, stream>>>(g_out_w, wouth, Fc * Fc);
  gemm_f16_kernel<false><<<dim3(F3c / 64, BSc / 128), gb, 0, stream>>>(
      xh, winh, g_in_b, nullptr, qkvh, BSc, F3c, Fc, F3c);
  flash_kernel<<<dim3(TILES / FW_WAVES), dim3(32 * FW_WAVES), 0, stream>>>(qkvh, attnh, 0);
  gemm_f16_kernel<true><<<dim3(Fc / 64, BSc / 128), gb, 0, stream>>>(
      attnh, wouth, g_out_b, projf, nullptr, BSc, Fc, Fc, 0);
  add_ln_kernel<<<BSc, cb, 0, stream>>>(x, projf, n2_w, n2_b, globf, gih + Fc, 2 * Fc);

  // ---- gate fusion ----
  gemm_f16_kernel<true><<<dim3(Fc / 64, BSc / 128), gb, 0, stream>>>(
      gih, gwh, gate_b, glinf, nullptr, BSc, Fc, 2 * Fc, 0);
  gate_fuse_kernel<<<BSc, cb, 0, stream>>>(glinf, gln_w, gln_b, localf, globf, (float*)d_out);
}
